// OpenSoraMultiHeadCrossAttention_10136122818878
// MI455X (gfx1250) — compile-verified, matrix operand order checked by reading, not run on hardware
//
#include <hip/hip_runtime.h>

// ---------------------------------------------------------------------------
// OpenSora MultiHeadCrossAttention for MI455X (gfx1250, wave32, WMMA bf16)
//
//   q  = x @ wq + bq                      (32768 x 1152)
//   kv = cond @ wkv + bkv                 (2400  x 2304)
//   s  = softmax(q kᵀ / sqrt(72) + mask)  per (b,h): (4096 x 300)
//   o  = s v ; out = o @ wp + bp          (32768 x 1152)
//
// All GEMMs on v_wmma_f32_16x16x32_bf16 (f32 accumulate). Global->LDS staging
// uses gfx1250 async loads (global_load_async_to_lds_b128 + s_wait_asynccnt),
// GEMM k-loop is double-buffered through LDS.
// ---------------------------------------------------------------------------

typedef __attribute__((ext_vector_type(16))) __bf16 v16bf;
typedef __attribute__((ext_vector_type(8)))  float  v8f;

union FragB16 {                 // one WMMA 16-bit operand (8 VGPRs)
    v16bf v;
    uint4 q[2];
    unsigned int u[8];
};

__device__ __forceinline__ unsigned short bf16_1(float f) {
    union { float f; unsigned int u; } x; x.f = f;
    unsigned int u = x.u + 0x7FFFu + ((x.u >> 16) & 1u);   // round-to-nearest-even
    return (unsigned short)(u >> 16);
}
__device__ __forceinline__ unsigned int pack2(float lo, float hi) {
    return (unsigned int)bf16_1(lo) | ((unsigned int)bf16_1(hi) << 16);
}
__device__ __forceinline__ v8f wmma_bf16(const FragB16& a, const FragB16& b, v8f c) {
    return __builtin_amdgcn_wmma_f32_16x16x32_bf16(false, a.v, false, b.v,
                                                   (short)0, c, false, false);
}

// --- gfx1250 async global->LDS (ASYNCcnt-tracked DMA into LDS) -------------
// Low 32 bits of a generic pointer to __shared__ == LDS byte offset
__device__ __forceinline__ void async_g2l_b128(void* lds_ptr, const void* gptr) {
    unsigned int l = (unsigned int)(uintptr_t)lds_ptr;
    asm volatile("global_load_async_to_lds_b128 %0, %1, off"
                 :: "v"(l), "v"(gptr) : "memory");
}
__device__ __forceinline__ void wait_async0() {
    asm volatile("s_wait_asynccnt 0x0" ::: "memory");
}

// ------------------------------ constants ----------------------------------
#define DM      1152
#define NHEADS  16
#define HD      72
#define HDP     96      // head dim padded to 3 k-steps of 32
#define MSEQ    300
#define MP      320     // kv length padded to 10 k-steps of 32 (20 n-tiles of 16)
#define DP      80      // head dim padded to 5 n-tiles of 16
#define BATCH   8
#define NSEQ    4096

// ---------------------- elementwise f32 -> bf16 ----------------------------
__global__ __launch_bounds__(256) void cvt_bf16_kernel(
        const float* __restrict__ in, unsigned int* __restrict__ out, long npairs) {
    long i = (long)blockIdx.x * 256 + threadIdx.x;
    if (i < npairs) {
        float2 f = ((const float2*)in)[i];
        out[i] = pack2(f.x, f.y);
    }
}

// ---------------- transpose+convert: out[c*R+r] = bf16(in[r*C+c]) ----------
__global__ __launch_bounds__(256) void transpose_bf16_kernel(
        const float* __restrict__ in, unsigned short* __restrict__ out, int R, int C) {
    long o = (long)blockIdx.x * 256 + threadIdx.x;
    if (o < (long)R * C) {
        int c = (int)(o / R), r = (int)(o % R);
        out[o] = bf16_1(in[(long)r * C + c]);
    }
}

// ------------------------------ GEMM ---------------------------------------
// C[M,N] = A[M,K] * Bt[N,K]^T + bias, K = 1152, block tile 128x128, 8 waves,
// double-buffered async-LDS pipeline over 36 k-steps of 32.
// EPI 0: bf16 row-major out.  EPI 1: f32 row-major out.  EPI 2: kv scatter.
#define LDW 40   // LDS row stride (ushorts) for 32-wide k tile: conflict-free

template <int EPI>
__global__ __launch_bounds__(256) void gemm_bf16_kernel(
        const unsigned short* __restrict__ A,
        const unsigned short* __restrict__ Bt,
        const float* __restrict__ bias,
        void* __restrict__ outp,
        unsigned short* __restrict__ kh,
        unsigned short* __restrict__ vt,
        int Mrows, int Ncols) {
    __shared__ unsigned short lA[2][128 * LDW];
    __shared__ unsigned short lB[2][128 * LDW];
    const int K = DM;
    const int KSTEPS = K / 32;   // 36

    int tid  = threadIdx.x;
    int lane = tid & 31, w = tid >> 5;
    int m0 = blockIdx.x * 128, n0 = blockIdx.y * 128;
    int wm = (w & 1) * 64, wn = (w >> 1) * 32;
    int r15 = lane & 15, hi = lane >> 4, kb = hi * 8;
    const uint4 z4 = make_uint4(0u, 0u, 0u, 0u);

    v8f acc[4][2];
#pragma unroll
    for (int i = 0; i < 4; ++i)
#pragma unroll
        for (int j = 0; j < 2; ++j)
#pragma unroll
            for (int e = 0; e < 8; ++e) acc[i][j][e] = 0.f;

    // async-stage one 128x32 A tile + 128x32 B tile into buffer `buf`
    auto stage = [&](int buf, int k0) {
#pragma unroll
        for (int c = tid; c < 512; c += 256) {
            int row = c >> 2, col8 = (c & 3) * 8;
            int gr = m0 + row;
            if (gr < Mrows)
                async_g2l_b128(&lA[buf][row * LDW + col8],
                               A + (size_t)gr * K + k0 + col8);
            else
                *(uint4*)(&lA[buf][row * LDW + col8]) = z4;
            async_g2l_b128(&lB[buf][row * LDW + col8],
                           Bt + (size_t)(n0 + row) * K + k0 + col8);
        }
    };

    stage(0, 0);
    wait_async0();
    __syncthreads();

    for (int kt = 0; kt < KSTEPS; ++kt) {
        int cur = kt & 1;
        if (kt + 1 < KSTEPS) stage(cur ^ 1, (kt + 1) * 32);

        FragB16 af[4], bfr[2];
#pragma unroll
        for (int i = 0; i < 4; ++i) {
            int r = wm + i * 16 + r15;
            af[i].q[0] = *(const uint4*)(&lA[cur][r * LDW + kb]);
            af[i].q[1] = *(const uint4*)(&lA[cur][r * LDW + 16 + kb]);
        }
#pragma unroll
        for (int j = 0; j < 2; ++j) {
            int r = wn + j * 16 + r15;
            bfr[j].q[0] = *(const uint4*)(&lB[cur][r * LDW + kb]);
            bfr[j].q[1] = *(const uint4*)(&lB[cur][r * LDW + 16 + kb]);
        }
#pragma unroll
        for (int i = 0; i < 4; ++i)
#pragma unroll
            for (int j = 0; j < 2; ++j)
                acc[i][j] = wmma_bf16(af[i], bfr[j], acc[i][j]);

        wait_async0();      // next tile resident
        __syncthreads();    // and this tile's readers done before overwrite
    }

    // epilogue: lane holds col r15, rows v + 8*hi of each 16x16 tile
#pragma unroll
    for (int i = 0; i < 4; ++i) {
#pragma unroll
        for (int j = 0; j < 2; ++j) {
            int col = n0 + wn + j * 16 + r15;
            float bv = bias[col];
#pragma unroll
            for (int v = 0; v < 8; ++v) {
                int row = m0 + wm + i * 16 + v + 8 * hi;
                if (row >= Mrows) continue;
                float val = acc[i][j][v] + bv;
                if (EPI == 0) {
                    ((unsigned short*)outp)[(size_t)row * Ncols + col] = bf16_1(val);
                } else if (EPI == 1) {
                    ((float*)outp)[(size_t)row * Ncols + col] = val;
                } else {  // kv scatter: K -> [b][h][m][d] pad, V -> [b][h][d][m] pad
                    int bb = row / MSEQ, mm = row % MSEQ;
                    if (col < DM) {
                        int h = col / HD, d = col % HD;
                        kh[(((size_t)bb * NHEADS + h) * MP + mm) * HDP + d] = bf16_1(val);
                    } else {
                        int c2 = col - DM;
                        int h = c2 / HD, d = c2 % HD;
                        vt[(((size_t)bb * NHEADS + h) * DP + d) * MP + mm] = bf16_1(val);
                    }
                }
            }
        }
    }
}

// --------------------------- attention -------------------------------------
// block = 8 waves; each wave owns a 16-row q tile; K/Vt shared in LDS.
#define K_STRIDE  104   // 96 + 8 pad  (bf16)  -> 52 dwords/row, conflict-free
#define VT_STRIDE 328   // 320 + 8 pad (bf16)  -> 164 dwords/row, conflict-free
#define SC_STRIDE 324   // 320 + 4 pad (f32)   -> half-waves 32 banks apart

__global__ __launch_bounds__(256) void attn_kernel(
        const unsigned short* __restrict__ qh,   // (B*N) x 1152 bf16
        const unsigned short* __restrict__ kh,   // [b][h][320][96] bf16 (zero padded)
        const unsigned short* __restrict__ vt,   // [b][h][80][320] bf16 (zero padded)
        const int* __restrict__ mask,
        unsigned short* __restrict__ oh) {       // (B*N) x 1152 bf16
    __shared__ unsigned short lK[MP * K_STRIDE];     //  66,560 B
    __shared__ unsigned short lV[DP * VT_STRIDE];    //  52,480 B
    __shared__ float          lS[8 * 16 * SC_STRIDE];// 165,888 B   (278 KB total)

    int tid = threadIdx.x, lane = tid & 31, w = tid >> 5;
    int bb = blockIdx.z, h = blockIdx.y;
    int n0 = blockIdx.x * 128 + w * 16;
    int maskb = mask[bb];
    const uint4 z4 = make_uint4(0u, 0u, 0u, 0u);

    // ---- async-stage K (320x96) and Vt (80x320) into LDS -------------------
    const unsigned short* kg = kh + ((size_t)bb * NHEADS + h) * MP * HDP;
    for (int c = tid; c < MP * (HDP / 8); c += 256) {             // 3840 chunks
        int row = c / 12, col8 = (c % 12) * 8;
        async_g2l_b128(&lK[row * K_STRIDE + col8], kg + row * HDP + col8);
    }
    for (int r = tid; r < MP; r += 256)
        *(uint4*)(&lK[r * K_STRIDE + HDP]) = z4;                  // pad 96..103
    const unsigned short* vg = vt + ((size_t)bb * NHEADS + h) * DP * MP;
    for (int c = tid; c < DP * (MP / 8); c += 256) {              // 3200 chunks
        int row = c / 40, col8 = (c % 40) * 8;
        async_g2l_b128(&lV[row * VT_STRIDE + col8], vg + row * MP + col8);
    }
    for (int r = tid; r < DP; r += 256)
        *(uint4*)(&lV[r * VT_STRIDE + MP]) = z4;                  // pad 320..327
    wait_async0();
    __syncthreads();

    int r15 = lane & 15, hi = lane >> 4, kb = hi * 8;

    // ---- q fragments (16 x 96, zero-padded beyond d=72) --------------------
    // ks=0/1 chunks are always inside the 72-wide head row; only ks=2 needs
    // masking.  Load unconditionally and cndmask the *value* (no pointer
    // select -> no flat/scratch path); the 16B over-read on the final row
    // stays inside the workspace (qh is followed by condh).
    FragB16 qa[3];
    {
        const unsigned short* qrow =
            qh + ((size_t)(bb * NSEQ) + n0 + r15) * DM + h * HD;
#pragma unroll
        for (int ks = 0; ks < 2; ++ks) {
            qa[ks].q[0] = *(const uint4*)(qrow + ks * 32 + kb);
            qa[ks].q[1] = *(const uint4*)(qrow + ks * 32 + 16 + kb);
        }
        uint4 t = *(const uint4*)(qrow + 64 + kb);   // d = 64 (valid) or 72 (pad)
        qa[2].q[0] = (kb == 0) ? t : z4;
        qa[2].q[1] = z4;
    }

    float* sw = &lS[w * 16 * SC_STRIDE];

    // ---- scores: 20 m-tiles x (3 WMMA) -> LDS ------------------------------
    for (int mt = 0; mt < MP / 16; ++mt) {
        v8f c = {0.f, 0.f, 0.f, 0.f, 0.f, 0.f, 0.f, 0.f};
#pragma unroll
        for (int ks = 0; ks < 3; ++ks) {
            FragB16 kf;
            const unsigned short* krow = &lK[(mt * 16 + r15) * K_STRIDE + ks * 32];
            kf.q[0] = *(const uint4*)(krow + kb);
            kf.q[1] = *(const uint4*)(krow + 16 + kb);
            c = wmma_bf16(qa[ks], kf, c);
        }
#pragma unroll
        for (int v = 0; v < 8; ++v)
            sw[(v + 8 * hi) * SC_STRIDE + mt * 16 + r15] = c[v];
    }

    // ---- masked softmax: 2 lanes per row, pair-combine via shfl_xor --------
    const float scale = 0.1178511301977579f;  // 1/sqrt(72)
    int srow = lane >> 1, shalf = lane & 1, mbase = shalf * 160;
    float* rp = &sw[srow * SC_STRIDE + mbase];
    float mx = -1e30f;
    for (int c = 0; c < 160; ++c)
        if (mbase + c < maskb) mx = fmaxf(mx, rp[c] * scale);
    mx = fmaxf(mx, __shfl_xor(mx, 1, 32));
    float s = 0.f;
    for (int c = 0; c < 160; ++c) {
        float e = 0.f;
        if (mbase + c < maskb) e = __expf(rp[c] * scale - mx);
        rp[c] = e;                    // also zeroes masked + padded columns
        s += e;
    }
    s += __shfl_xor(s, 1, 32);
    float rInv = 1.0f / __shfl(s, r15 * 2, 32);   // 1/sum for row r15

    // ---- PV: attn(16x320) @ Vt -> 5 d-tiles --------------------------------
    v8f acc[5];
#pragma unroll
    for (int dt = 0; dt < 5; ++dt)
#pragma unroll
        for (int e = 0; e < 8; ++e) acc[dt][e] = 0.f;

    for (int kt = 0; kt < MP / 32; ++kt) {
        FragB16 af;
        const float* arow = &sw[r15 * SC_STRIDE + kt * 32];
#pragma unroll
        for (int j = 0; j < 4; ++j)
            af.u[j] = pack2(arow[kb + 2 * j] * rInv, arow[kb + 2 * j + 1] * rInv);
#pragma unroll
        for (int j = 0; j < 4; ++j)
            af.u[4 + j] = pack2(arow[16 + kb + 2 * j] * rInv,
                                arow[16 + kb + 2 * j + 1] * rInv);
#pragma unroll
        for (int dt = 0; dt < 5; ++dt) {
            FragB16 vf;
            const unsigned short* vrow = &lV[(dt * 16 + r15) * VT_STRIDE + kt * 32];
            vf.q[0] = *(const uint4*)(vrow + kb);
            vf.q[1] = *(const uint4*)(vrow + 16 + kb);
            acc[dt] = wmma_bf16(af, vf, acc[dt]);
        }
    }

    // ---- store o (bf16, [b][n][h*72+d]) ------------------------------------
#pragma unroll
    for (int dt = 0; dt < 5; ++dt) {
        int d = dt * 16 + r15;
        if (d < HD) {
#pragma unroll
            for (int v = 0; v < 8; ++v) {
                int n = n0 + v + 8 * hi;
                oh[((size_t)(bb * NSEQ) + n) * DM + h * HD + d] = bf16_1(acc[dt][v]);
            }
        }
    }
}

// ------------------------------ launch -------------------------------------
extern "C" void kernel_launch(void* const* d_in, const int* in_sizes, int n_in,
                              void* d_out, int out_size, void* d_ws, size_t ws_size,
                              hipStream_t stream) {
    (void)in_sizes; (void)n_in; (void)out_size; (void)ws_size;
    const float* x    = (const float*)d_in[0];
    const float* cond = (const float*)d_in[1];
    const int*   mask = (const int*)d_in[2];
    const float* wq   = (const float*)d_in[3];
    const float* bq   = (const float*)d_in[4];
    const float* wkv  = (const float*)d_in[5];
    const float* bkv  = (const float*)d_in[6];
    const float* wp   = (const float*)d_in[7];
    const float* bp   = (const float*)d_in[8];

    const size_t NX = (size_t)BATCH * NSEQ * DM;   // 37,748,736
    const size_t NC = (size_t)BATCH * MSEQ * DM;   //  2,764,800

    char* ws = (char*)d_ws;
    size_t off = 0;
    auto take = [&](size_t bytes) { char* p = ws + off; off += (bytes + 255) & ~255ull; return p; };
    unsigned short* xh    = (unsigned short*)take(NX * 2);  // also reused as `oh`
    unsigned short* qh    = (unsigned short*)take(NX * 2);
    unsigned short* condh = (unsigned short*)take(NC * 2);
    unsigned short* wqT   = (unsigned short*)take((size_t)DM * DM * 2);
    unsigned short* wkvT  = (unsigned short*)take((size_t)2 * DM * DM * 2);
    unsigned short* wpT   = (unsigned short*)take((size_t)DM * DM * 2);
    size_t khBytes = (size_t)BATCH * NHEADS * MP * HDP * 2;   // 7,864,320
    size_t vtBytes = (size_t)BATCH * NHEADS * DP * MP * 2;    // 6,553,600
    unsigned short* kh = (unsigned short*)take(khBytes);
    unsigned short* vt = (unsigned short*)take(vtBytes);      // contiguous after kh

    // 1) precision conversion + weight transposes
    cvt_bf16_kernel<<<(unsigned)((NX / 2 + 255) / 256), 256, 0, stream>>>(x, (unsigned int*)xh, (long)(NX / 2));
    cvt_bf16_kernel<<<(unsigned)((NC / 2 + 255) / 256), 256, 0, stream>>>(cond, (unsigned int*)condh, (long)(NC / 2));
    long nw = (long)DM * DM;
    transpose_bf16_kernel<<<(unsigned)((nw + 255) / 256), 256, 0, stream>>>(wq, wqT, DM, DM);
    transpose_bf16_kernel<<<(unsigned)((2 * nw + 255) / 256), 256, 0, stream>>>(wkv, wkvT, DM, 2 * DM);
    transpose_bf16_kernel<<<(unsigned)((nw + 255) / 256), 256, 0, stream>>>(wp, wpT, DM, DM);
    hipMemsetAsync(kh, 0, khBytes + vtBytes, stream);         // zero k/v padding

    // 2) q = x @ wq + bq  -> bf16
    gemm_bf16_kernel<0><<<dim3(256, 9), 256, 0, stream>>>(
        xh, wqT, bq, qh, nullptr, nullptr, BATCH * NSEQ, DM);

    // 3) kv = cond @ wkv + bkv -> scatter into padded K / Vt
    gemm_bf16_kernel<2><<<dim3(19, 18), 256, 0, stream>>>(
        condh, wkvT, bkv, nullptr, kh, vt, BATCH * MSEQ, 2 * DM);

    // 4) attention (writes o into xh, which is dead after step 2)
    attn_kernel<<<dim3(NSEQ / 128, NHEADS, BATCH), 256, 0, stream>>>(
        qh, kh, vt, mask, xh);

    // 5) out = o @ wp + bp -> f32
    gemm_bf16_kernel<1><<<dim3(256, 9), 256, 0, stream>>>(
        xh, wpT, bp, d_out, nullptr, nullptr, BATCH * NSEQ, DM);
}
